// BigramLM_34273839022823
// MI455X (gfx1250) — compile-verified, matrix operand order checked by reading, not run on hardware
//
#include <hip/hip_runtime.h>
#include <math.h>

// ---------------------------------------------------------------------------
// Types
// ---------------------------------------------------------------------------
typedef _Float16 half_t;
typedef __attribute__((ext_vector_type(16))) _Float16 v16h;
typedef __attribute__((ext_vector_type(8)))  _Float16 v8h;
typedef __attribute__((ext_vector_type(8)))  float    v8f;

// Model constants
#define BB   64
#define TT   256
#define EE   384
#define HH   6
#define HSS  64
#define LL   10
#define VV   65
#define FFD  1536
#define NTOK (BB * TT)

// ---------------------------------------------------------------------------
// WMMA helpers (CDNA5: v_wmma_f32_16x16x32_f16, wave32)
// ---------------------------------------------------------------------------
__device__ __forceinline__ v8f wmma16(v16h a, v16h b, v8f c) {
  return __builtin_amdgcn_wmma_f32_16x16x32_f16(
      /*neg_a=*/false, a, /*neg_b=*/false, b,
      /*c_mod=*/(short)0, c, /*reuse_a=*/false, /*reuse_b=*/false);
}

// Load one A/B fragment for 16x16x32 f16 WMMA (ISA 7.12.2 layout):
// lane&15 selects row (A) / col (B); kb = (lane>=16)*8.
// Elements 0..7  <- K = kb..kb+7 ; elements 8..15 <- K = 16+kb..16+kb+7.
// rowbase must be 16B aligned.
__device__ __forceinline__ v16h load_frag(const half_t* rowbase, int kb) {
  v8h lo = *reinterpret_cast<const v8h*>(rowbase + kb);
  v8h hi = *reinterpret_cast<const v8h*>(rowbase + kb + 16);
  v16h r;
#pragma unroll
  for (int i = 0; i < 8; ++i) { r[i] = lo[i]; r[i + 8] = hi[i]; }
  return r;
}

// ---------------------------------------------------------------------------
// Weight packing: f32 [L][K][N] -> f16 transposed [L][N][K] (once per launch)
// ---------------------------------------------------------------------------
__global__ void pack_transpose_f16(const float* __restrict__ src,
                                   half_t* __restrict__ dst,
                                   int Lc, int Kc, int Nc) {
  const int idx = blockIdx.x * blockDim.x + threadIdx.x;
  if (idx >= Lc * Kc * Nc) return;
  const int l = idx / (Kc * Nc);
  const int r = idx % (Kc * Nc);
  const int k = r / Nc;
  const int n = r % Nc;
  dst[(size_t)l * Kc * Nc + (size_t)n * Kc + k] = (half_t)src[idx];
}

// ---------------------------------------------------------------------------
// Embedding: x = tok_emb[enc] + pos_emb
// ---------------------------------------------------------------------------
__global__ void embed_kernel(const int* __restrict__ enc,
                             const float* __restrict__ tok,
                             const float* __restrict__ pos,
                             float* __restrict__ X) {
  const int idx = blockIdx.x * blockDim.x + threadIdx.x;
  if (idx >= NTOK * EE) return;
  const int n = idx / EE;
  const int e = idx % EE;
  X[idx] = tok[(size_t)enc[n] * EE + e] + pos[(size_t)(n & (TT - 1)) * EE + e];
}

// ---------------------------------------------------------------------------
// LayerNorm: f32 in -> f16 out. One wave per row (32 lanes x 12 elems, E=384).
// ---------------------------------------------------------------------------
__global__ __launch_bounds__(256) void layernorm_f16(
    const float* __restrict__ X, const float* __restrict__ gamma,
    const float* __restrict__ beta, half_t* __restrict__ Y) {
  const int lane = threadIdx.x & 31;
  const int w    = threadIdx.x >> 5;
  const int row  = blockIdx.x * 8 + w;
  const float* xr = X + (size_t)row * EE;
  float v[12], s = 0.f;
#pragma unroll
  for (int i = 0; i < 12; ++i) { v[i] = xr[i * 32 + lane]; s += v[i]; }
#pragma unroll
  for (int msk = 1; msk < 32; msk <<= 1) s += __shfl_xor(s, msk, 32);
  const float mu = s * (1.0f / (float)EE);
  float s2 = 0.f;
#pragma unroll
  for (int i = 0; i < 12; ++i) { float d = v[i] - mu; s2 += d * d; }
#pragma unroll
  for (int msk = 1; msk < 32; msk <<= 1) s2 += __shfl_xor(s2, msk, 32);
  const float rstd = rsqrtf(s2 * (1.0f / (float)EE) + 1e-5f);
  half_t* yr = Y + (size_t)row * EE;
#pragma unroll
  for (int i = 0; i < 12; ++i) {
    const int c = i * 32 + lane;
    yr[c] = (half_t)((v[i] - mu) * rstd * gamma[c] + beta[c]);
  }
}

// ---------------------------------------------------------------------------
// f16 WMMA GEMM: C[M,N] = act(A[M,K] @ WT[N,K]^T + bias) (+resid)
// Block tile 128x128, 8 waves, K-step 32. A tile staged via
// GLOBAL_LOAD_ASYNC_TO_LDS_B128 (ASYNCcnt); WT tile via b128 copies.
// Epilogue: f32 / f16 row-major / f16 transposed (vectorized) outputs.
// ---------------------------------------------------------------------------
#define GF_BIAS  1
#define GF_RESID 2
#define GF_RELU  4
#define GF_OUTF  8
#define GF_OUTH  16
#define GF_OUTHT 32

__global__ __launch_bounds__(256) void gemm_f16_wmma(
    const half_t* __restrict__ A, const half_t* __restrict__ WT,
    const float* __restrict__ bias, const float* __restrict__ resid,
    float* __restrict__ outF, half_t* __restrict__ outH,
    half_t* __restrict__ outHT,
    int M, int N, int K, int flags) {
  constexpr int LS = 40;  // LDS row stride in halves (80B: 16B-aligned rows)
  __shared__ __attribute__((aligned(16))) half_t lA[128 * LS];  // [row][k]
  __shared__ __attribute__((aligned(16))) half_t lW[128 * LS];  // [col][k]

  const int tid    = threadIdx.x;
  const int lane   = tid & 31;
  const int wave   = tid >> 5;        // 8 row tiles of 16
  const int m0     = blockIdx.y * 128;
  const int n0     = blockIdx.x * 128;
  const int lane15 = lane & 15;
  const int kb     = (lane >> 4) << 3;
  const int hadd   = kb;

  v8f c[8] = {};

  const int ldRow = tid >> 1;         // 0..127
  const int ldCol = (tid & 1) * 16;   // 0 or 16

  const unsigned laDst = (unsigned)(uintptr_t)(&lA[ldRow * LS + ldCol]);
  const bool wRowOk = (n0 + ldRow) < N;  // Wout tail rows zero-filled
  const half_t* wRowPtr = WT + (size_t)(n0 + ldRow) * K + ldCol;

  for (int k0 = 0; k0 < K; k0 += 32) {
    // --- A tile: async global -> LDS (16B per lane per op, offset hits both) ---
    const half_t* aSrc = A + (size_t)(m0 + ldRow) * K + k0 + ldCol;
    asm volatile("global_load_async_to_lds_b128 %0, %1, off"
                 :: "v"(laDst), "v"(aSrc) : "memory");
    asm volatile("global_load_async_to_lds_b128 %0, %1, off offset:16"
                 :: "v"(laDst), "v"(aSrc) : "memory");
    if (k0 + 32 < K) __builtin_prefetch(aSrc + 32, 0, 1);

    // --- W^T tile: vector copy (row-guarded for the N=65 logits tail) ---
    v8h w0 = {}, w1 = {};
    if (wRowOk) {
      w0 = *reinterpret_cast<const v8h*>(wRowPtr + k0);
      w1 = *reinterpret_cast<const v8h*>(wRowPtr + k0 + 8);
    }
    *reinterpret_cast<v8h*>(&lW[ldRow * LS + ldCol])     = w0;
    *reinterpret_cast<v8h*>(&lW[ldRow * LS + ldCol + 8]) = w1;

    asm volatile("s_wait_asynccnt 0" ::: "memory");
    __syncthreads();

    const v16h aF = load_frag(&lA[(wave * 16 + lane15) * LS], kb);
#pragma unroll
    for (int nt = 0; nt < 8; ++nt) {
      const v16h bF = load_frag(&lW[(nt * 16 + lane15) * LS], kb);
      c[nt] = wmma16(aF, bF, c[nt]);
    }
    __syncthreads();
  }

  // Epilogue (C layout: lane&15 = col, element r -> row r + 8*(lane>=16))
  const int rowBase = m0 + wave * 16 + hadd;
#pragma unroll
  for (int nt = 0; nt < 8; ++nt) {
    const int col = n0 + nt * 16 + lane15;
    if (col >= N) continue;
    float vv[8];
#pragma unroll
    for (int r = 0; r < 8; ++r) {
      float v = c[nt][r];
      if (flags & GF_BIAS)  v += bias[col];
      if (flags & GF_RELU)  v = fmaxf(v, 0.f);
      if (flags & GF_RESID) v += resid[(size_t)(rowBase + r) * N + col];
      vv[r] = v;
    }
    if (flags & GF_OUTF) {
#pragma unroll
      for (int r = 0; r < 8; ++r) outF[(size_t)(rowBase + r) * N + col] = vv[r];
    }
    if (flags & GF_OUTH) {
#pragma unroll
      for (int r = 0; r < 8; ++r) outH[(size_t)(rowBase + r) * N + col] = (half_t)vv[r];
    }
    if (flags & GF_OUTHT) {  // transposed: 8 consecutive rows -> one v8h store
      v8h ov;
#pragma unroll
      for (int r = 0; r < 8; ++r) ov[r] = (half_t)vv[r];
      *reinterpret_cast<v8h*>(outHT + (size_t)col * M + rowBase) = ov;
    }
  }
}

// ---------------------------------------------------------------------------
// Flash attention: 4 waves x 16 query rows per block, online softmax over
// 32-key blocks, causal, scale = E^-0.5 (matches reference). Q/K row-major
// f16 [NTOK,E]; V pre-transposed f16 [E][NTOK] so P@V B-frags load straight
// from global. Only P goes through LDS (C-layout -> A-layout transpose).
// No cross-wave barriers (causal per-wave trip counts).
// ---------------------------------------------------------------------------
__global__ __launch_bounds__(128) void attention_wmma(
    const half_t* __restrict__ Q, const half_t* __restrict__ Kt,
    const half_t* __restrict__ VT, half_t* __restrict__ O, float scale) {
  constexpr int PS = 40;
  __shared__ __attribute__((aligned(16))) half_t pt[4][16 * PS];  // P [row][key]

  const int lane   = threadIdx.x & 31;
  const int w      = threadIdx.x >> 5;
  const int bh     = blockIdx.x;
  const int b      = bh / HH, h = bh % HH;
  const int qbase  = blockIdx.y * 64 + w * 16;
  const int lane15 = lane & 15;
  const int kb     = (lane >> 4) << 3;
  const int hadd   = kb;
  const size_t headOff = (size_t)h * HSS;
  const int tokBase = b * TT;

  // Q fragments (A layout), K-dim = HS split in two 32-chunks
  const half_t* qp = Q + ((size_t)(tokBase + qbase + lane15)) * EE + headOff;
  const v16h qA0 = load_frag(qp, kb);
  const v16h qA1 = load_frag(qp + 32, kb);

  v8f o0 = {}, o1 = {}, o2 = {}, o3 = {};
  float m[8], l[8];
#pragma unroll
  for (int r = 0; r < 8; ++r) { m[r] = -1e30f; l[r] = 0.f; }

  const int nblk = (qbase + 47) >> 5;  // causal: keys up to qbase+15
  for (int blk = 0; blk < nblk; ++blk) {
    const int kbase = blk * 32;

    // Scores: S[16q x 32k] = Q @ K^T (two 16-key N-tiles, two HS chunks)
    v8f s0 = {}, s1 = {};
    {
      const half_t* kp0 = Kt + ((size_t)(tokBase + kbase + lane15)) * EE + headOff;
      const half_t* kp1 = kp0 + (size_t)16 * EE;
      v16h kf;
      kf = load_frag(kp0, kb);      s0 = wmma16(qA0, kf, s0);
      kf = load_frag(kp0 + 32, kb); s0 = wmma16(qA1, kf, s0);
      kf = load_frag(kp1, kb);      s1 = wmma16(qA0, kf, s1);
      kf = load_frag(kp1 + 32, kb); s1 = wmma16(qA1, kf, s1);
    }

    // Online softmax update; write P (f16) to LDS in A-matrix order
#pragma unroll
    for (int r = 0; r < 8; ++r) {
      const int row  = qbase + r + hadd;
      const int col0 = kbase + lane15;
      float x0 = s0[r] * scale; if (col0 > row)      x0 = -1e30f;
      float x1 = s1[r] * scale; if (col0 + 16 > row) x1 = -1e30f;
      float bm = fmaxf(x0, x1);
#pragma unroll
      for (int msk = 1; msk < 16; msk <<= 1) bm = fmaxf(bm, __shfl_xor(bm, msk, 32));
      const float mn = fmaxf(m[r], bm);
      const float sc = __expf(m[r] - mn);
      const float p0 = __expf(x0 - mn);
      const float p1 = __expf(x1 - mn);
      float rs = p0 + p1;
#pragma unroll
      for (int msk = 1; msk < 16; msk <<= 1) rs += __shfl_xor(rs, msk, 32);
      m[r] = mn;
      l[r] = l[r] * sc + rs;
      o0[r] *= sc; o1[r] *= sc; o2[r] *= sc; o3[r] *= sc;
      pt[w][(r + hadd) * PS + lane15]      = (half_t)p0;
      pt[w][(r + hadd) * PS + 16 + lane15] = (half_t)p1;
    }

    // attn += P @ V : B-frags straight from V^T global [E][NTOK]
    const v16h pA = load_frag(&pt[w][lane15 * PS], kb);
    const half_t* vb = VT + (headOff + lane15) * (size_t)NTOK + tokBase + kbase;
    v16h vB;
    vB = load_frag(vb,                         kb); o0 = wmma16(pA, vB, o0);
    vB = load_frag(vb + (size_t)16 * NTOK,     kb); o1 = wmma16(pA, vB, o1);
    vB = load_frag(vb + (size_t)32 * NTOK,     kb); o2 = wmma16(pA, vB, o2);
    vB = load_frag(vb + (size_t)48 * NTOK,     kb); o3 = wmma16(pA, vB, o3);
  }

  // Normalize and store f16 row-major (feeds Wo GEMM as A)
#pragma unroll
  for (int r = 0; r < 8; ++r) {
    const float inv = 1.0f / l[r];
    const int row = qbase + r + hadd;
    half_t* op = O + ((size_t)(tokBase + row)) * EE + headOff;
    op[lane15]      = (half_t)(o0[r] * inv);
    op[16 + lane15] = (half_t)(o1[r] * inv);
    op[32 + lane15] = (half_t)(o2[r] * inv);
    op[48 + lane15] = (half_t)(o3[r] * inv);
  }
}

// ---------------------------------------------------------------------------
// Host orchestration
// ---------------------------------------------------------------------------
extern "C" void kernel_launch(void* const* d_in, const int* in_sizes, int n_in,
                              void* d_out, int out_size, void* d_ws, size_t ws_size,
                              hipStream_t stream) {
  (void)in_sizes; (void)n_in; (void)out_size; (void)ws_size;
  const int*   enc  = (const int*)  d_in[0];
  const float* tok  = (const float*)d_in[1];
  const float* pos  = (const float*)d_in[2];
  const float* Wq   = (const float*)d_in[3];
  const float* Wk   = (const float*)d_in[4];
  const float* Wv   = (const float*)d_in[5];
  const float* Wo   = (const float*)d_in[6];
  const float* bo   = (const float*)d_in[7];
  const float* W1   = (const float*)d_in[8];
  const float* b1   = (const float*)d_in[9];
  const float* W2   = (const float*)d_in[10];
  const float* b2   = (const float*)d_in[11];
  const float* ln1s = (const float*)d_in[12];
  const float* ln1b = (const float*)d_in[13];
  const float* ln2s = (const float*)d_in[14];
  const float* ln2b = (const float*)d_in[15];
  const float* lnfs = (const float*)d_in[16];
  const float* lnfb = (const float*)d_in[17];
  const float* Wout = (const float*)d_in[18];
  const float* bout = (const float*)d_in[19];

  // Workspace carve-up (deterministic)
  char* ws = (char*)d_ws;
  size_t off = 0;
  auto alloc = [&](size_t bytes) -> void* {
    void* p = ws + off;
    off = (off + bytes + 255) & ~(size_t)255;
    return p;
  };
  // All weights stored transposed: [N][K] f16
  half_t* wqT  = (half_t*)alloc((size_t)LL * EE * EE * 2);
  half_t* wkT  = (half_t*)alloc((size_t)LL * EE * EE * 2);
  half_t* wvT  = (half_t*)alloc((size_t)LL * EE * EE * 2);
  half_t* woT  = (half_t*)alloc((size_t)LL * EE * EE * 2);
  half_t* w1T  = (half_t*)alloc((size_t)LL * EE * FFD * 2);
  half_t* w2T  = (half_t*)alloc((size_t)LL * FFD * EE * 2);
  half_t* woutT = (half_t*)alloc((size_t)VV * EE * 2);
  float*  X    = (float*) alloc((size_t)NTOK * EE * 4);
  half_t* h16  = (half_t*)alloc((size_t)NTOK * EE * 2);
  half_t* q16  = (half_t*)alloc((size_t)NTOK * EE * 2);
  half_t* k16  = (half_t*)alloc((size_t)NTOK * EE * 2);
  half_t* vT16 = (half_t*)alloc((size_t)NTOK * EE * 2);   // [E][NTOK]
  half_t* a16  = (half_t*)alloc((size_t)NTOK * EE * 2);
  half_t* ff16 = (half_t*)alloc((size_t)NTOK * FFD * 2);

  // --- weight packing (transpose to [N][K] f16) ---
  {
    // Wq/Wk/Wv: [L,H,E,HS] -> per (l,h): transpose [E][HS] -> rows h*HS+d.
    // Contiguity makes this exactly generic transpose with L'=L*H, K=E, N=HS.
    const int nq = LL * HH * EE * HSS;
    pack_transpose_f16<<<(nq + 255) / 256, 256, 0, stream>>>(Wq, wqT, LL * HH, EE, HSS);
    pack_transpose_f16<<<(nq + 255) / 256, 256, 0, stream>>>(Wk, wkT, LL * HH, EE, HSS);
    pack_transpose_f16<<<(nq + 255) / 256, 256, 0, stream>>>(Wv, wvT, LL * HH, EE, HSS);
    const int nwo = LL * EE * EE;
    pack_transpose_f16<<<(nwo + 255) / 256, 256, 0, stream>>>(Wo, woT, LL, EE, EE);
    const int nw1 = LL * EE * FFD;
    pack_transpose_f16<<<(nw1 + 255) / 256, 256, 0, stream>>>(W1, w1T, LL, EE, FFD);
    pack_transpose_f16<<<(nw1 + 255) / 256, 256, 0, stream>>>(W2, w2T, LL, FFD, EE);
    const int nwv = EE * VV;
    pack_transpose_f16<<<(nwv + 255) / 256, 256, 0, stream>>>(Wout, woutT, 1, EE, VV);
  }

  // --- embedding ---
  embed_kernel<<<(NTOK * EE + 255) / 256, 256, 0, stream>>>(enc, tok, pos, X);

  const dim3 gemmBlk(256);
  const dim3 gE(EE / 128, NTOK / 128);    // N=384
  const dim3 gF(FFD / 128, NTOK / 128);   // N=1536
  const dim3 gV(1, NTOK / 128);           // N=65 (one col block)
  const dim3 lnGrid(NTOK / 8);
  const dim3 attGrid(BB * HH, TT / 64);
  const float scale = 1.0f / sqrtf((float)EE);  // reference scales by E^-0.5

  for (int ly = 0; ly < LL; ++ly) {
    const size_t wEE = (size_t)ly * EE * EE;
    // LN1 -> h16 (f16)
    layernorm_f16<<<lnGrid, 256, 0, stream>>>(X, ln1s + ly * EE, ln1b + ly * EE, h16);
    // QKV projections (no bias); V written transposed
    gemm_f16_wmma<<<gE, gemmBlk, 0, stream>>>(
        h16, wqT + wEE, nullptr, nullptr, nullptr, q16, nullptr,
        NTOK, EE, EE, GF_OUTH);
    gemm_f16_wmma<<<gE, gemmBlk, 0, stream>>>(
        h16, wkT + wEE, nullptr, nullptr, nullptr, k16, nullptr,
        NTOK, EE, EE, GF_OUTH);
    gemm_f16_wmma<<<gE, gemmBlk, 0, stream>>>(
        h16, wvT + wEE, nullptr, nullptr, nullptr, nullptr, vT16,
        NTOK, EE, EE, GF_OUTHT);
    // Attention
    attention_wmma<<<attGrid, 128, 0, stream>>>(q16, k16, vT16, a16, scale);
    // x = x + attn @ Wo + bo
    gemm_f16_wmma<<<gE, gemmBlk, 0, stream>>>(
        a16, woT + wEE, bo + ly * EE, X, X, nullptr, nullptr,
        NTOK, EE, EE, GF_BIAS | GF_RESID | GF_OUTF);
    // LN2 -> h16
    layernorm_f16<<<lnGrid, 256, 0, stream>>>(X, ln2s + ly * EE, ln2b + ly * EE, h16);
    // ff = relu(h @ W1 + b1)
    gemm_f16_wmma<<<gF, gemmBlk, 0, stream>>>(
        h16, w1T + (size_t)ly * EE * FFD, b1 + ly * FFD, nullptr, nullptr, ff16,
        nullptr, NTOK, FFD, EE, GF_BIAS | GF_RELU | GF_OUTH);
    // x = x + ff @ W2 + b2
    gemm_f16_wmma<<<gE, gemmBlk, 0, stream>>>(
        ff16, w2T + (size_t)ly * FFD * EE, b2 + ly * EE, X, X, nullptr, nullptr,
        NTOK, EE, FFD, GF_BIAS | GF_RESID | GF_OUTF);
  }

  // Final LN + logits (N=65: W^T row guard zero-fills, stores col-guarded)
  layernorm_f16<<<lnGrid, 256, 0, stream>>>(X, lnfs, lnfb, h16);
  gemm_f16_wmma<<<gV, gemmBlk, 0, stream>>>(
      h16, woutT, bout, nullptr, (float*)d_out, nullptr, nullptr,
      NTOK, VV, EE, GF_BIAS | GF_OUTF);
}
